// MoeDecoderLayerPP_47802986004941
// MI455X (gfx1250) — matrix operand order repaired and checked by reading, not compiled, not run on hardware
//
#include <hip/hip_runtime.h>
#include <hip/hip_bf16.h>
#include <math.h>

// ---------------- model dims (compile-time) ----------------
#define Bb   2
#define Ss   2048
#define Dd   1024
#define Hh   16
#define KVh  8
#define HDd  64
#define Ee   8
#define Kk   2
#define FFf  3584
#define Nn   (Bb*Ss)            // 4096 tokens
#define QD   (Hh*HDd)           // 1024
#define KD   (KVh*HDd)          // 512
#define NSLOT (Nn*Kk + 16*Ee)   // 8320 padded routed slots
#define NT   (NSLOT/16)         // 520 row tiles
#define FT   (FFf/16)           // 224 ff tiles
#define DT   (Dd/16)            // 64 d tiles

typedef _Float16 half_t;
typedef half_t v16h __attribute__((ext_vector_type(16)));
typedef half_t v8h  __attribute__((ext_vector_type(8)));
typedef float  v8f  __attribute__((ext_vector_type(8)));

#define DEV static __device__ __forceinline__

// D = A(16x32 f16) * B(32x16 f16) + C(16x16 f32), wave32 WMMA
DEV v8f wmma_f16(v16h a, v16h b, v8f c) {
  return __builtin_amdgcn_wmma_f32_16x16x32_f16(false, a, false, b, (short)0, c,
                                                false, false);
}

// A fragment (16x32, f16). p = row_base + (lane>>4)*16. Elements:
//   k = k0 + (lane>>4)*16 + e  -> two contiguous 16B loads.
DEV v16h ldA(const half_t* p, int k0) {
  v8h lo = *(const v8h*)(p + k0);
  v8h hi = *(const v8h*)(p + k0 + 8);
  v16h r;
#pragma unroll
  for (int i = 0; i < 8; ++i) { r[i] = lo[i]; r[i + 8] = hi[i]; }
  return r;
}

// B fragment (32x16, f16) from TRANSPOSED weight [N][K].
// p = (n0 + lane&15)*K + (lane>>4)*16 : elements k = k0 + (lane>>4)*16 + e
DEV v16h ldBT(const half_t* p, int k0) {
  v8h lo = *(const v8h*)(p + k0);
  v8h hi = *(const v8h*)(p + k0 + 8);
  v16h r;
#pragma unroll
  for (int i = 0; i < 8; ++i) { r[i] = lo[i]; r[i + 8] = hi[i]; }
  return r;
}

// B fragment from ROW-MAJOR K x N matrix (strided): element e -> row +e
DEV v16h ldB_rm(const half_t* p, int ld) {
  v16h r;
#pragma unroll
  for (int e = 0; e < 16; ++e) r[e] = p[(size_t)e * ld];
  return r;
}

// ---------------- utility kernels ----------------
__global__ void zero_kernel(unsigned int* p, int nwords) {
  int i = blockIdx.x * blockDim.x + threadIdx.x;
  if (i < nwords) p[i] = 0u;
}

// batched f32[rows][cols] -> f16 transposed [cols][rows]
__global__ void convT_kernel(const float* __restrict__ src,
                             half_t* __restrict__ dst, int rows, int cols) {
  size_t bo = (size_t)blockIdx.y * rows * cols;
  size_t tot = (size_t)rows * cols;
  for (size_t i = blockIdx.x * (size_t)blockDim.x + threadIdx.x; i < tot;
       i += (size_t)gridDim.x * blockDim.x) {
    int r = (int)(i / cols), c = (int)(i % cols);
    dst[bo + (size_t)c * rows + r] = (half_t)src[bo + i];
  }
}

// ---------------- RMSNorm ----------------
__global__ void rmsnorm_kernel(const float* __restrict__ x,
                               const float* __restrict__ w,
                               half_t* __restrict__ yh, float* __restrict__ yf) {
  int t = blockIdx.x, tid = threadIdx.x;               // 256 threads / token
  const float* row = x + (size_t)t * Dd;
  float ss = 0.f;
  for (int d = tid; d < Dd; d += 256) { float v = row[d]; ss += v * v; }
  __shared__ float red[256];
  red[tid] = ss; __syncthreads();
  for (int s = 128; s > 0; s >>= 1) {
    if (tid < s) red[tid] += red[tid + s];
    __syncthreads();
  }
  float inv = rsqrtf(red[0] / (float)Dd + 1e-5f);
  for (int d = tid; d < Dd; d += 256) {
    float v = row[d] * inv * w[d];
    yh[(size_t)t * Dd + d] = (half_t)v;
    if (yf) yf[(size_t)t * Dd + d] = v;
  }
}

// ---- generic GEMM, 16x64 strip/wave: C[M][N] = A*BT (+resid), f32 out ------
__global__ void __launch_bounds__(32)
gemm_kernel(const half_t* __restrict__ A, const half_t* __restrict__ BT,
            float* __restrict__ C, const float* __restrict__ resid,
            int Kd, int Ncols) {
  int n0 = blockIdx.x * 64, m0 = blockIdx.y * 16;
  int lane = threadIdx.x, li = lane & 15, hs = lane >> 4;
  const half_t* ap = A + (size_t)(m0 + li) * Kd + hs * 16;
  const half_t* bp0 = BT + (size_t)(n0 + li) * Kd + hs * 16;
  const half_t* bp1 = bp0 + (size_t)16 * Kd;
  const half_t* bp2 = bp0 + (size_t)32 * Kd;
  const half_t* bp3 = bp0 + (size_t)48 * Kd;
  v8f c0 = {}, c1 = {}, c2 = {}, c3 = {};
  for (int k = 0; k < Kd; k += 32) {
    __builtin_prefetch(bp0 + k + 256, 0, 1);
    __builtin_prefetch(bp2 + k + 256, 0, 1);
    v16h a = ldA(ap, k);
    c0 = wmma_f16(a, ldBT(bp0, k), c0);
    c1 = wmma_f16(a, ldBT(bp1, k), c1);
    c2 = wmma_f16(a, ldBT(bp2, k), c2);
    c3 = wmma_f16(a, ldBT(bp3, k), c3);
  }
#pragma unroll
  for (int r = 0; r < 8; ++r) {
    size_t row = (size_t)(m0 + r + 8 * hs) * Ncols + n0 + li;
    float v0 = c0[r], v1 = c1[r], v2 = c2[r], v3 = c3[r];
    if (resid) {
      v0 += resid[row]; v1 += resid[row + 16];
      v2 += resid[row + 32]; v3 += resid[row + 48];
    }
    C[row] = v0; C[row + 16] = v1; C[row + 32] = v2; C[row + 48] = v3;
  }
}

// ---- QKV GEMM, one 64-wide head strip/wave, fused RoPE + f16 store ---------
// RoPE pairing d <-> d+-32 lands on (c0<->c2) and (c1<->c3) in registers.
__global__ void __launch_bounds__(32)
gemm_qkv_kernel(const half_t* __restrict__ A, const half_t* __restrict__ BT,
                const int* __restrict__ posids, half_t* __restrict__ outh,
                int Kd, int Ncols, int do_rope) {
  int n0 = blockIdx.x * 64, m0 = blockIdx.y * 16;
  int lane = threadIdx.x, li = lane & 15, hs = lane >> 4;
  const half_t* ap = A + (size_t)(m0 + li) * Kd + hs * 16;
  const half_t* bp0 = BT + (size_t)(n0 + li) * Kd + hs * 16;
  const half_t* bp1 = bp0 + (size_t)16 * Kd;
  const half_t* bp2 = bp0 + (size_t)32 * Kd;
  const half_t* bp3 = bp0 + (size_t)48 * Kd;
  v8f c0 = {}, c1 = {}, c2 = {}, c3 = {};
  for (int k = 0; k < Kd; k += 32) {
    __builtin_prefetch(bp0 + k + 256, 0, 1);
    __builtin_prefetch(bp2 + k + 256, 0, 1);
    v16h a = ldA(ap, k);
    c0 = wmma_f16(a, ldBT(bp0, k), c0);
    c1 = wmma_f16(a, ldBT(bp1, k), c1);
    c2 = wmma_f16(a, ldBT(bp2, k), c2);
    c3 = wmma_f16(a, ldBT(bp3, k), c3);
  }
  // inv_freq for this lane's two column phases (d%32 = li and li+16)
  float if0 = powf(10000.f, -((float)(2 * li)) / (float)HDd);
  float if1 = powf(10000.f, -((float)(2 * (li + 16))) / (float)HDd);
#pragma unroll
  for (int r = 0; r < 8; ++r) {
    int m = m0 + r + 8 * hs;                       // token index
    float v0 = c0[r], v1 = c1[r], v2 = c2[r], v3 = c3[r];
    if (do_rope) {
      float pos = (float)posids[m];
      float a0 = pos * if0, a1 = pos * if1;
      float cs0 = cosf(a0), sn0 = sinf(a0);
      float cs1 = cosf(a1), sn1 = sinf(a1);
      float o0 = v0 * cs0 - v2 * sn0;              // d = li
      float o2 = v2 * cs0 + v0 * sn0;              // d = li+32
      float o1 = v1 * cs1 - v3 * sn1;              // d = li+16
      float o3 = v3 * cs1 + v1 * sn1;              // d = li+48
      v0 = o0; v1 = o1; v2 = o2; v3 = o3;
    }
    size_t row = (size_t)m * Ncols + n0 + li;
    outh[row]      = (half_t)v0;
    outh[row + 16] = (half_t)v1;
    outh[row + 32] = (half_t)v2;
    outh[row + 48] = (half_t)v3;
  }
}

// ---------------- flash attention, one wave per (b, h, 16-query tile) -------
__global__ void __launch_bounds__(32)
attn_kernel(const half_t* __restrict__ qh, const half_t* __restrict__ kh,
            const half_t* __restrict__ vh, half_t* __restrict__ oh) {
  const int qt = blockIdx.x, h = blockIdx.y, b = blockIdx.z;
  const int lane = threadIdx.x, li = lane & 15, hs = lane >> 4;
  const int kvh = h / (Hh / KVh);
  const int qbase = qt * 16;
  const size_t tok0 = (size_t)b * Ss;

  __shared__ float  sS[16][32];
  __shared__ half_t sP[16][32];
  __shared__ float  sMax[16], sSum[16], sAlpha[16];
  if (lane < 16) { sMax[lane] = -3.0e38f; sSum[lane] = 0.f; }
  __syncthreads();

  const half_t* qp = qh + (tok0 + qbase + li) * QD + h * HDd + hs * 16;
  v16h q0 = ldA(qp, 0);
  v16h q1 = ldA(qp, 32);
  v8f o0 = {}, o1 = {}, o2 = {}, o3 = {};

  const int kend = qbase + 16;                       // exclusive causal limit
  for (int kb = 0; kb < kend; kb += 32) {
    // ---- S = Q * K^T (two 16x16 score tiles over this 32-key chunk) ----
    v8f s0 = {}, s1 = {};
    {
      const half_t* kp0 = kh + (tok0 + kb + li) * KD + kvh * HDd + hs * 16;
      s0 = wmma_f16(q0, ldBT(kp0, 0),  s0);
      s0 = wmma_f16(q1, ldBT(kp0, 32), s0);
      const half_t* kp1 = kp0 + (size_t)16 * KD;
      s1 = wmma_f16(q0, ldBT(kp1, 0),  s1);
      s1 = wmma_f16(q1, ldBT(kp1, 32), s1);
    }
#pragma unroll
    for (int r = 0; r < 8; ++r) {
      int m = r + 8 * hs;
      int qidx = qbase + m;
      sS[m][li]      = (kb + li      <= qidx) ? s0[r] * 0.125f : -3.0e38f;
      sS[m][16 + li] = (kb + 16 + li <= qidx) ? s1[r] * 0.125f : -3.0e38f;
    }
    __syncthreads();
    if (lane < 16) {                                  // per-row online softmax
      int m = lane;
      float mx = sMax[m];
#pragma unroll
      for (int j = 0; j < 32; ++j) mx = fmaxf(mx, sS[m][j]);
      float alpha = expf(sMax[m] - mx);
      float sum = sSum[m] * alpha;
#pragma unroll
      for (int j = 0; j < 32; ++j) {
        float p = expf(sS[m][j] - mx);
        sP[m][j] = (half_t)p;
        sum += p;
      }
      sMax[m] = mx; sSum[m] = sum; sAlpha[m] = alpha;
    }
    __syncthreads();
    {
      float al[8];
#pragma unroll
      for (int r = 0; r < 8; ++r) al[r] = sAlpha[r + 8 * hs];
#pragma unroll
      for (int r = 0; r < 8; ++r) {
        o0[r] *= al[r]; o1[r] *= al[r]; o2[r] *= al[r]; o3[r] *= al[r];
      }
      v16h pa = ldA(&sP[li][0] + hs * 16, 0);          // P as A fragment
      const half_t* vp = vh + (tok0 + kb + hs * 16) * KD + kvh * HDd + li;
      o0 = wmma_f16(pa, ldB_rm(vp + 0,  KD), o0);
      o1 = wmma_f16(pa, ldB_rm(vp + 16, KD), o1);
      o2 = wmma_f16(pa, ldB_rm(vp + 32, KD), o2);
      o3 = wmma_f16(pa, ldB_rm(vp + 48, KD), o3);
    }
    __syncthreads();
  }
#pragma unroll
  for (int r = 0; r < 8; ++r) {
    int m = r + 8 * hs;
    float inv = 1.f / sSum[m];
    size_t base = (tok0 + qbase + m) * (size_t)QD + h * HDd;
    oh[base + 0  + li] = (half_t)(o0[r] * inv);
    oh[base + 16 + li] = (half_t)(o1[r] * inv);
    oh[base + 32 + li] = (half_t)(o2[r] * inv);
    oh[base + 48 + li] = (half_t)(o3[r] * inv);
  }
}

// ---------------- router: logits, softmax, top-2, stats ----------------
__global__ void gate_kernel(const float* __restrict__ xf,
                            const float* __restrict__ Wg,
                            int* __restrict__ top_i, float* __restrict__ top_w,
                            float* __restrict__ cnt, float* __restrict__ probsum) {
  int t = blockIdx.x, tid = threadIdx.x;              // 128 threads / token
  float part[Ee];
#pragma unroll
  for (int e = 0; e < Ee; ++e) part[e] = 0.f;
  const float* row = xf + (size_t)t * Dd;
  for (int d = tid; d < Dd; d += 128) {
    float xv = row[d];
#pragma unroll
    for (int e = 0; e < Ee; ++e) part[e] += xv * Wg[d * Ee + e];
  }
  __shared__ float sacc[Ee];
  if (tid < Ee) sacc[tid] = 0.f;
  __syncthreads();
#pragma unroll
  for (int e = 0; e < Ee; ++e) atomicAdd(&sacc[e], part[e]);
  __syncthreads();
  if (tid == 0) {
    float p[Ee], mx = -3e38f;
    for (int e = 0; e < Ee; ++e) mx = fmaxf(mx, sacc[e]);
    float sum = 0.f;
    for (int e = 0; e < Ee; ++e) { p[e] = expf(sacc[e] - mx); sum += p[e]; }
    float inv = 1.f / sum;
    int i0 = 0;
    for (int e = 1; e < Ee; ++e) if (p[e] > p[i0]) i0 = e;
    int i1 = (i0 == 0) ? 1 : 0;
    for (int e = 0; e < Ee; ++e) if (e != i0 && p[e] > p[i1]) i1 = e;
    float ws = p[i0] + p[i1];
    top_i[2 * t] = i0;             top_i[2 * t + 1] = i1;
    top_w[2 * t] = p[i0] / ws;     top_w[2 * t + 1] = p[i1] / ws;
    atomicAdd(&cnt[i0], 1.f);      atomicAdd(&cnt[Ee + i1], 1.f);
    for (int e = 0; e < Ee; ++e) atomicAdd(&probsum[e], p[e] * inv);
  }
}

// padded per-expert offsets, tile->expert map, load-balance loss
__global__ void route_finalize_kernel(const float* __restrict__ cnt,
                                      const float* __restrict__ probsum,
                                      const float* __restrict__ lb_in,
                                      int* __restrict__ off,
                                      int* __restrict__ tile_expert,
                                      float* __restrict__ out_lb) {
  int o = 0; off[0] = 0;
  for (int e = 0; e < Ee; ++e) {
    int c = (int)(cnt[e] + 0.5f) + (int)(cnt[Ee + e] + 0.5f);
    c = (c + 15) & ~15;
    o += c; off[e + 1] = o;
  }
  for (int tIdx = 0; tIdx < NT; ++tIdx) {
    int row = tIdx * 16, te = -1;
    for (int e = 0; e < Ee; ++e)
      if (row >= off[e] && row < off[e + 1]) te = e;
    tile_expert[tIdx] = te;
  }
  float lb = 0.f;
  for (int k = 0; k < Kk; ++k) {
    float acc = 0.f;
    for (int e = 0; e < Ee; ++e)
      acc += (cnt[k * Ee + e] / (float)Nn) * (probsum[e] / (float)Nn);
    lb += acc;
  }
  lb = lb / (float)Kk * (float)Ee;
  *out_lb = lb_in[0] + lb;
}

__global__ void scatter_kernel(const int* __restrict__ top_i,
                               const float* __restrict__ top_w,
                               const int* __restrict__ off,
                               int* __restrict__ fill, int* __restrict__ perm,
                               float* __restrict__ wslot) {
  int t = blockIdx.x * blockDim.x + threadIdx.x;
  if (t >= Nn) return;
  for (int k = 0; k < Kk; ++k) {
    int e = top_i[2 * t + k];
    int pos = off[e] + atomicAdd(&fill[e], 1);
    perm[pos] = t;
    wslot[pos] = top_w[2 * t + k];
  }
}

// ---- MoE pass 1: g = silu(x@w1) * (x@w3); 2 ff-tiles x (w1,w3) per wave ----
__global__ void __launch_bounds__(32)
moe_ffn1_kernel(const half_t* __restrict__ xh, const int* __restrict__ perm,
                const int* __restrict__ tile_expert,
                const half_t* __restrict__ w1T, const half_t* __restrict__ w3T,
                half_t* __restrict__ g) {
  int tf0 = blockIdx.x * 2, tr = blockIdx.y;
  int e = tile_expert[tr];
  if (e < 0) return;
  int lane = threadIdx.x, li = lane & 15, hs = lane >> 4;
  int tok = perm[tr * 16 + li];
  const half_t* ap = xh + (size_t)tok * Dd + hs * 16;
  size_t wbase = (size_t)e * FFf * Dd + (size_t)(tf0 * 16 + li) * Dd + hs * 16;
  const half_t* b1a = w1T + wbase;
  const half_t* b1b = b1a + (size_t)16 * Dd;
  const half_t* b3a = w3T + wbase;
  const half_t* b3b = b3a + (size_t)16 * Dd;
  v8f c1a = {}, c1b = {}, c3a = {}, c3b = {};
  for (int k = 0; k < Dd; k += 32) {
    __builtin_prefetch(b1a + k + 256, 0, 1);
    __builtin_prefetch(b3a + k + 256, 0, 1);
    v16h a = ldA(ap, k);
    c1a = wmma_f16(a, ldBT(b1a, k), c1a);
    c1b = wmma_f16(a, ldBT(b1b, k), c1b);
    c3a = wmma_f16(a, ldBT(b3a, k), c3a);
    c3b = wmma_f16(a, ldBT(b3b, k), c3b);
  }
#pragma unroll
  for (int r = 0; r < 8; ++r) {
    int m = r + 8 * hs;
    size_t row = (size_t)(tr * 16 + m) * FFf + tf0 * 16 + li;
    float h1 = c1a[r], h3 = c3a[r];
    g[row] = (half_t)((h1 / (1.f + expf(-h1))) * h3);
    h1 = c1b[r]; h3 = c3b[r];
    g[row + 16] = (half_t)((h1 / (1.f + expf(-h1))) * h3);
  }
}

// ---- MoE pass 2: out[tok] += w * (g @ w2); 16x64 strip per wave ----
__global__ void __launch_bounds__(32)
moe_ffn2_kernel(const half_t* __restrict__ g, const half_t* __restrict__ w2T,
                const int* __restrict__ perm, const float* __restrict__ wslot,
                const int* __restrict__ tile_expert, float* __restrict__ out) {
  int d0 = blockIdx.x * 64, tr = blockIdx.y;
  int e = tile_expert[tr];
  if (e < 0) return;
  int lane = threadIdx.x, li = lane & 15, hs = lane >> 4;
  const half_t* ap = g + (size_t)(tr * 16 + li) * FFf + hs * 16;
  const half_t* bp0 = w2T + (size_t)e * Dd * FFf + (size_t)(d0 + li) * FFf + hs * 16;
  const half_t* bp1 = bp0 + (size_t)16 * FFf;
  const half_t* bp2 = bp0 + (size_t)32 * FFf;
  const half_t* bp3 = bp0 + (size_t)48 * FFf;
  v8f c0 = {}, c1 = {}, c2 = {}, c3 = {};
  for (int k = 0; k < FFf; k += 32) {
    __builtin_prefetch(bp0 + k + 512, 0, 1);
    __builtin_prefetch(bp2 + k + 512, 0, 1);
    v16h a = ldA(ap, k);
    c0 = wmma_f16(a, ldBT(bp0, k), c0);
    c1 = wmma_f16(a, ldBT(bp1, k), c1);
    c2 = wmma_f16(a, ldBT(bp2, k), c2);
    c3 = wmma_f16(a, ldBT(bp3, k), c3);
  }
#pragma unroll
  for (int r = 0; r < 8; ++r) {
    int slot = tr * 16 + r + 8 * hs;
    float w = wslot[slot];
    if (w != 0.f) {
      float* orow = out + (size_t)perm[slot] * Dd + d0 + li;
      atomicAdd(orow,      c0[r] * w);
      atomicAdd(orow + 16, c1[r] * w);
      atomicAdd(orow + 32, c2[r] * w);
      atomicAdd(orow + 48, c3[r] * w);
    }
  }
}

// ---------------- host orchestration ----------------
extern "C" void kernel_launch(void* const* d_in, const int* in_sizes, int n_in,
                              void* d_out, int out_size, void* d_ws, size_t ws_size,
                              hipStream_t stream) {
  (void)in_sizes; (void)n_in; (void)out_size; (void)ws_size;
  const float* hidden = (const float*)d_in[0];
  const int*   posids = (const int*)d_in[1];
  const float* lb_in  = (const float*)d_in[2];
  const float* ln1_w  = (const float*)d_in[3];
  const float* ln2_w  = (const float*)d_in[4];
  const float* Wq     = (const float*)d_in[5];
  const float* Wk     = (const float*)d_in[6];
  const float* Wv     = (const float*)d_in[7];
  const float* Wo     = (const float*)d_in[8];
  const float* Wg     = (const float*)d_in[9];
  const float* w1     = (const float*)d_in[10];
  const float* w2     = (const float*)d_in[11];
  const float* w3     = (const float*)d_in[12];
  float* out = (float*)d_out;                  // [Nn*Dd] hidden + [1] lb

  // ---- workspace carve-out (256B aligned slabs) ----
  char* p = (char*)d_ws;
  auto alloc = [&](size_t bytes) -> void* {
    void* r = (void*)p;
    p += (bytes + 255) & ~(size_t)255;
    return r;
  };
  half_t* xn1   = (half_t*)alloc((size_t)Nn * Dd * 2);
  half_t* xn2   = (half_t*)alloc((size_t)Nn * Dd * 2);
  float*  xn2f  = (float*)alloc((size_t)Nn * Dd * 4);
  half_t* WqT   = (half_t*)alloc((size_t)QD * Dd * 2);
  half_t* WkT   = (half_t*)alloc((size_t)KD * Dd * 2);
  half_t* WvT   = (half_t*)alloc((size_t)KD * Dd * 2);
  half_t* WoT   = (half_t*)alloc((size_t)Dd * QD * 2);
  half_t* w1T   = (half_t*)alloc((size_t)Ee * FFf * Dd * 2);
  half_t* w3T   = (half_t*)alloc((size_t)Ee * FFf * Dd * 2);
  half_t* w2T   = (half_t*)alloc((size_t)Ee * Dd * FFf * 2);
  half_t* qhb   = (half_t*)alloc((size_t)Nn * QD * 2);
  half_t* khb   = (half_t*)alloc((size_t)Nn * KD * 2);
  half_t* vhb   = (half_t*)alloc((size_t)Nn * KD * 2);
  half_t* attnh = (half_t*)alloc((size_t)Nn * QD * 2);
  half_t* gbuf  = (half_t*)alloc((size_t)NSLOT * FFf * 2);
  // routing block zeroed each call: cnt[2*E] | probsum[E] | fill[E] | perm | wslot
  const int ZWORDS = 2 * Ee + Ee + Ee + NSLOT + NSLOT;
  unsigned int* zbase = (unsigned int*)alloc((size_t)ZWORDS * 4);
  float* cnt     = (float*)zbase;
  float* probsum = cnt + 2 * Ee;
  int*   fill    = (int*)(probsum + Ee);
  int*   perm    = fill + Ee;
  float* wslot   = (float*)(perm + NSLOT);
  int*   off     = (int*)alloc(16 * 4);
  int*   texp    = (int*)alloc((size_t)NT * 4);
  int*   top_i   = (int*)alloc((size_t)Nn * Kk * 4);
  float* top_w   = (float*)alloc((size_t)Nn * Kk * 4);

  // ---- 1. weight conversion to transposed f16 ----
  convT_kernel<<<dim3(512, 1), 256, 0, stream>>>(Wq, WqT, Dd, QD);
  convT_kernel<<<dim3(512, 1), 256, 0, stream>>>(Wk, WkT, Dd, KD);
  convT_kernel<<<dim3(512, 1), 256, 0, stream>>>(Wv, WvT, Dd, KD);
  convT_kernel<<<dim3(512, 1), 256, 0, stream>>>(Wo, WoT, QD, Dd);
  convT_kernel<<<dim3(1024, Ee), 256, 0, stream>>>(w1, w1T, Dd, FFf);
  convT_kernel<<<dim3(1024, Ee), 256, 0, stream>>>(w3, w3T, Dd, FFf);
  convT_kernel<<<dim3(1024, Ee), 256, 0, stream>>>(w2, w2T, FFf, Dd);

  // ---- 2. attention block ----
  rmsnorm_kernel<<<Nn, 256, 0, stream>>>(hidden, ln1_w, xn1, nullptr);
  // QKV with fused RoPE + f16 output (64-wide strips = one head per wave)
  gemm_qkv_kernel<<<dim3(QD / 64, Nn / 16), 32, 0, stream>>>(
      xn1, WqT, posids, qhb, Dd, QD, 1);
  gemm_qkv_kernel<<<dim3(KD / 64, Nn / 16), 32, 0, stream>>>(
      xn1, WkT, posids, khb, Dd, KD, 1);
  gemm_qkv_kernel<<<dim3(KD / 64, Nn / 16), 32, 0, stream>>>(
      xn1, WvT, posids, vhb, Dd, KD, 0);
  attn_kernel<<<dim3(Ss / 16, Hh, Bb), 32, 0, stream>>>(qhb, khb, vhb, attnh);
  // out = hidden + attn @ Wo   (residual fused in epilogue)
  gemm_kernel<<<dim3(Dd / 64, Nn / 16), 32, 0, stream>>>(attnh, WoT, out,
                                                         hidden, QD, Dd);

  // ---- 3. MoE block ----
  rmsnorm_kernel<<<Nn, 256, 0, stream>>>(out, ln2_w, xn2, xn2f);
  zero_kernel<<<(ZWORDS + 255) / 256, 256, 0, stream>>>(zbase, ZWORDS);
  gate_kernel<<<Nn, 128, 0, stream>>>(xn2f, Wg, top_i, top_w, cnt, probsum);
  route_finalize_kernel<<<1, 1, 0, stream>>>(cnt, probsum, lb_in, off, texp,
                                             out + (size_t)Nn * Dd);
  scatter_kernel<<<(Nn + 255) / 256, 256, 0, stream>>>(top_i, top_w, off, fill,
                                                       perm, wslot);
  moe_ffn1_kernel<<<dim3(FT / 2, NT), 32, 0, stream>>>(xn2, perm, texp, w1T,
                                                       w3T, gbuf);
  moe_ffn2_kernel<<<dim3(DT / 4, NT), 32, 0, stream>>>(gbuf, w2T, perm, wslot,
                                                       texp, out);
}
